// LatticeRNNCell_91268055040455
// MI455X (gfx1250) — compile-verified, matrix operand order checked by reading, not run on hardware
//
#include <hip/hip_runtime.h>

// ---------------------------------------------------------------------------
// Problem constants
// ---------------------------------------------------------------------------
#define BATCH 4096
#define INSZ  512
#define HID   1024
#define K3H   3072      // 3*HID
#define KCAT  1536      // INSZ + HID
#define N4H   4096      // 4*HID

// GEMM tiling
#define BM 128
#define BN 128
#define BK 32
#define LDSK 40         // LDS row pitch in bf16 elements (80 bytes, 16B aligned)
#define TILE_ELEMS (BM * LDSK)   // one staged tile (A or W)

typedef __attribute__((ext_vector_type(16))) __bf16 bf16x16;
typedef __attribute__((ext_vector_type(8)))  float  f32x8;

struct alignas(16) U128 { unsigned int x, y, z, w; };
struct alignas(8)  U64  { unsigned int x, y; };
struct alignas(16) F128 { float x, y, z, w; };

__device__ __forceinline__ unsigned short f2bf(float f) {
    unsigned int u = __float_as_uint(f);
    u += 0x7FFFu + ((u >> 16) & 1u);          // round-to-nearest-even
    return (unsigned short)(u >> 16);
}
__device__ __forceinline__ unsigned int pack2(float a, float b) {
    return (unsigned int)f2bf(a) | ((unsigned int)f2bf(b) << 16);
}

// ---------------------------------------------------------------------------
// CDNA5 async global->LDS copy (ASYNCcnt-tracked, bypasses VGPRs).
// LDS destination operand is the 32-bit LDS byte offset: for a generic pointer
// into __shared__, the LDS offset is the low 32 bits of the address, so truncation is OK.
// ---------------------------------------------------------------------------
__device__ __forceinline__ unsigned lds_off32(const void* p) {
    return (unsigned)(unsigned long long)p;
}
__device__ __forceinline__ void async_copy_b128(unsigned lds_byte_off, const void* gsrc) {
    asm volatile("global_load_async_to_lds_b128 %0, %1, off"
                 :: "v"(lds_byte_off), "v"(gsrc)
                 : "memory");
}
__device__ __forceinline__ void wait_async0() {
    asm volatile("s_wait_asynccnt 0x0" ::: "memory");
}

// ---------------------------------------------------------------------------
// Conversion / packing kernels (4 elements per thread, all sizes divide 4)
// ---------------------------------------------------------------------------
__global__ __launch_bounds__(256) void f32_to_bf16_kernel(const float* __restrict__ src,
                                                          unsigned short* __restrict__ dst,
                                                          int n) {
    int j = (blockIdx.x * 256 + threadIdx.x) * 4;
    if (j >= n) return;
    F128 f = *(const F128*)(src + j);
    U64 o; o.x = pack2(f.x, f.y); o.y = pack2(f.z, f.w);
    *(U64*)(dst + j) = o;
}

// Wcat[n][0:512) = W_ih[n][:], Wcat[n][512:1536) = W_hh[n][:]   (bf16, [4096 x 1536])
__global__ __launch_bounds__(256) void pack_wcat_kernel(const float* __restrict__ Wih,
                                                        const float* __restrict__ Whh,
                                                        unsigned short* __restrict__ dst) {
    int j = (blockIdx.x * 256 + threadIdx.x) * 4;
    int row = j / KCAT, col = j % KCAT;            // chunk of 4 never crosses col 512
    const float* s = (col < INSZ) ? (Wih + (size_t)row * INSZ + col)
                                  : (Whh + (size_t)row * HID + (col - INSZ));
    F128 f = *(const F128*)s;
    U64 o; o.x = pack2(f.x, f.y); o.y = pack2(f.z, f.w);
    *(U64*)(dst + (size_t)row * KCAT + col) = o;
}

// Xcat[m][0:512) = bf16(x[m][:])  (cols 512:1536 filled later by LayerNorm-h)
__global__ __launch_bounds__(256) void pack_x_kernel(const float* __restrict__ x,
                                                     unsigned short* __restrict__ dst) {
    int j = (blockIdx.x * 256 + threadIdx.x) * 4;
    int row = j / INSZ, col = j % INSZ;
    F128 f = *(const F128*)(x + (size_t)row * INSZ + col);
    U64 o; o.x = pack2(f.x, f.y); o.y = pack2(f.z, f.w);
    *(U64*)(dst + (size_t)row * KCAT + col) = o;
}

// ---------------------------------------------------------------------------
// WMMA fragment helpers
// 16-bit A-matrix 16x32 layout (ISA 7.12.2): lane<16 -> K 0..7 & 16..23,
// lane>=16 -> K 8..15 & 24..31, row = lane&15.  B fragments use the same
// pattern on W^T rows (weights are [N,K] row-major already).
// ---------------------------------------------------------------------------
__device__ __forceinline__ bf16x16 frag_load(const unsigned short* tile, int rowBase, int lane) {
    const int lrow = lane & 15;
    const int koff = (lane & 16) ? 8 : 0;
    const unsigned short* p = tile + (rowBase + lrow) * LDSK + koff;
    union { bf16x16 v; U128 u[2]; } f;
    f.u[0] = *(const U128*)(p);
    f.u[1] = *(const U128*)(p + 16);
    return f.v;
}

__device__ __forceinline__ void mma_step(const unsigned short* At, const unsigned short* Wt,
                                         int waveM, int waveN, int lane, f32x8 acc[4][2]) {
    bf16x16 a[4], b[2];
#pragma unroll
    for (int mt = 0; mt < 4; ++mt) a[mt] = frag_load(At, waveM * 64 + mt * 16, lane);
#pragma unroll
    for (int nt = 0; nt < 2; ++nt) b[nt] = frag_load(Wt, waveN * 32 + nt * 16, lane);
#pragma unroll
    for (int mt = 0; mt < 4; ++mt)
#pragma unroll
        for (int nt = 0; nt < 2; ++nt)
            acc[mt][nt] = __builtin_amdgcn_wmma_f32_16x16x32_bf16(
                false, a[mt], false, b[nt], (short)0, acc[mt][nt], false, false);
}

// ---------------------------------------------------------------------------
// GEMM 1/2: raw[M=4096, N=1024] = concat(s0,s1,s2)[M,3072] @ Wbf[N,3072]^T
// A: fp32 -> bf16 VALU convert into LDS.  W: async global->LDS b128 copies.
// ---------------------------------------------------------------------------
__global__ __launch_bounds__(256) void gemm_proc_kernel(const float* __restrict__ s0,
                                                        const float* __restrict__ s1,
                                                        const float* __restrict__ s2,
                                                        const unsigned short* __restrict__ Wbf,
                                                        float* __restrict__ out) {
    __shared__ unsigned short smem[4 * TILE_ELEMS];   // 40 KB: [A0][W0][A1][W1]

    const int tid    = threadIdx.x;
    const int mBlock = blockIdx.y * BM;
    const int nBlock = blockIdx.x * BN;
    const int wid    = tid >> 5, lane = tid & 31;
    const int waveM  = wid >> 2, waveN = wid & 3;
    const int row    = tid >> 1, kh = tid & 1;       // loader: 128 rows x 2 half-K

    f32x8 acc[4][2];
    const f32x8 zero = {0.f, 0.f, 0.f, 0.f, 0.f, 0.f, 0.f, 0.f};
#pragma unroll
    for (int mt = 0; mt < 4; ++mt)
#pragma unroll
        for (int nt = 0; nt < 2; ++nt) acc[mt][nt] = zero;

    auto loadA = [&](unsigned short* dst, int k0) {
        const int gk = k0 + kh * 16;                 // 16-aligned, never crosses HID boundary
        const float* src;
        if      (gk < HID)     src = s0 + (size_t)(mBlock + row) * HID + gk;
        else if (gk < 2 * HID) src = s1 + (size_t)(mBlock + row) * HID + (gk - HID);
        else                   src = s2 + (size_t)(mBlock + row) * HID + (gk - 2 * HID);
        F128 f0 = *(const F128*)(src);
        F128 f1 = *(const F128*)(src + 4);
        F128 f2 = *(const F128*)(src + 8);
        F128 f3 = *(const F128*)(src + 12);
        U128 p0, p1;
        p0.x = pack2(f0.x, f0.y); p0.y = pack2(f0.z, f0.w);
        p0.z = pack2(f1.x, f1.y); p0.w = pack2(f1.z, f1.w);
        p1.x = pack2(f2.x, f2.y); p1.y = pack2(f2.z, f2.w);
        p1.z = pack2(f3.x, f3.y); p1.w = pack2(f3.z, f3.w);
        *(U128*)(dst + row * LDSK + kh * 16)     = p0;
        *(U128*)(dst + row * LDSK + kh * 16 + 8) = p1;
    };
    auto loadW = [&](unsigned short* dst, int k0) {
        const unsigned short* p = Wbf + (size_t)(nBlock + row) * K3H + k0 + kh * 16;
        const unsigned d0 = lds_off32(dst + row * LDSK + kh * 16);
        async_copy_b128(d0,      p);
        async_copy_b128(d0 + 16, p + 8);
    };

    loadA(smem, 0);
    loadW(smem + TILE_ELEMS, 0);
    wait_async0();
    __syncthreads();

    const int KSTEPS = K3H / BK;                     // 96
    for (int ks = 0; ks < KSTEPS; ++ks) {
        const int cur = ks & 1;
        unsigned short* Ac = smem + cur * (2 * TILE_ELEMS);
        unsigned short* Wc = Ac + TILE_ELEMS;
        unsigned short* An = smem + (cur ^ 1) * (2 * TILE_ELEMS);
        unsigned short* Wn = An + TILE_ELEMS;
        if (ks + 1 < KSTEPS) { loadW(Wn, (ks + 1) * BK); loadA(An, (ks + 1) * BK); }
        mma_step(Ac, Wc, waveM, waveN, lane, acc);
        wait_async0();
        __syncthreads();
    }

    const int lcol  = lane & 15;
    const int rhalf = (lane & 16) ? 8 : 0;           // C/D layout: lanes 16..31 hold M+8
#pragma unroll
    for (int mt = 0; mt < 4; ++mt) {
        const int gr0 = mBlock + waveM * 64 + mt * 16 + rhalf;
#pragma unroll
        for (int nt = 0; nt < 2; ++nt) {
            const int gc = nBlock + waveN * 32 + nt * 16 + lcol;
#pragma unroll
            for (int r = 0; r < 8; ++r)
                out[(size_t)(gr0 + r) * HID + gc] = acc[mt][nt][r];
        }
    }
}

// ---------------------------------------------------------------------------
// GEMM 3: gates[4096, 4096] = Xcat[4096,1536](bf16) @ Wcat[4096,1536]^T(bf16)
// Both tiles staged with async global->LDS b128 copies.
// ---------------------------------------------------------------------------
__global__ __launch_bounds__(256) void gemm_gates_kernel(const unsigned short* __restrict__ Abf,
                                                         const unsigned short* __restrict__ Wbf,
                                                         float* __restrict__ out) {
    __shared__ unsigned short smem[4 * TILE_ELEMS];

    const int tid    = threadIdx.x;
    const int mBlock = blockIdx.y * BM;
    const int nBlock = blockIdx.x * BN;
    const int wid    = tid >> 5, lane = tid & 31;
    const int waveM  = wid >> 2, waveN = wid & 3;
    const int row    = tid >> 1, kh = tid & 1;

    f32x8 acc[4][2];
    const f32x8 zero = {0.f, 0.f, 0.f, 0.f, 0.f, 0.f, 0.f, 0.f};
#pragma unroll
    for (int mt = 0; mt < 4; ++mt)
#pragma unroll
        for (int nt = 0; nt < 2; ++nt) acc[mt][nt] = zero;

    auto loadT = [&](unsigned short* dst, const unsigned short* src, int rBase, int k0) {
        const unsigned short* p = src + (size_t)(rBase + row) * KCAT + k0 + kh * 16;
        const unsigned d0 = lds_off32(dst + row * LDSK + kh * 16);
        async_copy_b128(d0,      p);
        async_copy_b128(d0 + 16, p + 8);
    };

    loadT(smem, Abf, mBlock, 0);
    loadT(smem + TILE_ELEMS, Wbf, nBlock, 0);
    wait_async0();
    __syncthreads();

    const int KSTEPS = KCAT / BK;                    // 48
    for (int ks = 0; ks < KSTEPS; ++ks) {
        const int cur = ks & 1;
        unsigned short* Ac = smem + cur * (2 * TILE_ELEMS);
        unsigned short* Wc = Ac + TILE_ELEMS;
        unsigned short* An = smem + (cur ^ 1) * (2 * TILE_ELEMS);
        unsigned short* Wn = An + TILE_ELEMS;
        if (ks + 1 < KSTEPS) {
            loadT(An, Abf, mBlock, (ks + 1) * BK);
            loadT(Wn, Wbf, nBlock, (ks + 1) * BK);
        }
        mma_step(Ac, Wc, waveM, waveN, lane, acc);
        wait_async0();
        __syncthreads();
    }

    const int lcol  = lane & 15;
    const int rhalf = (lane & 16) ? 8 : 0;
#pragma unroll
    for (int mt = 0; mt < 4; ++mt) {
        const int gr0 = mBlock + waveM * 64 + mt * 16 + rhalf;
#pragma unroll
        for (int nt = 0; nt < 2; ++nt) {
            const int gc = nBlock + waveN * 32 + nt * 16 + lcol;
#pragma unroll
            for (int r = 0; r < 8; ++r)
                out[(size_t)(gr0 + r) * N4H + gc] = acc[mt][nt][r];
        }
    }
}

// ---------------------------------------------------------------------------
// LayerNorm over rows of 1024: y = (x+bias - mu) * rsqrt(var+eps) * gamma + beta
// One 256-thread block (8 waves) per row; 4 cols/thread.
// h-path: write bf16 into Xcat cols 512..1535.  c-path: write fp32 proc_c.
// ---------------------------------------------------------------------------
__global__ __launch_bounds__(256) void ln_kernel(const float* __restrict__ raw,
                                                 const float* __restrict__ bias,
                                                 const float* __restrict__ gamma,
                                                 const float* __restrict__ beta,
                                                 float* __restrict__ out_f32,
                                                 unsigned short* __restrict__ out_bf16) {
    __shared__ float ssum[8], ssq[8], sres[2];
    const int row = blockIdx.x;
    const int tid = threadIdx.x;
    const int c   = tid * 4;

    F128 v = *(const F128*)(raw + (size_t)row * HID + c);
    F128 b = *(const F128*)(bias + c);
    float x0 = v.x + b.x, x1 = v.y + b.y, x2 = v.z + b.z, x3 = v.w + b.w;
    float s = x0 + x1 + x2 + x3;
    float q = x0 * x0 + x1 * x1 + x2 * x2 + x3 * x3;
#pragma unroll
    for (int off = 16; off > 0; off >>= 1) {
        s += __shfl_xor(s, off, 32);
        q += __shfl_xor(q, off, 32);
    }
    if ((tid & 31) == 0) { ssum[tid >> 5] = s; ssq[tid >> 5] = q; }
    __syncthreads();
    if (tid == 0) {
        float S = 0.f, Q = 0.f;
#pragma unroll
        for (int i = 0; i < 8; ++i) { S += ssum[i]; Q += ssq[i]; }
        float mu  = S * (1.0f / HID);
        float var = Q * (1.0f / HID) - mu * mu;
        sres[0] = mu;
        sres[1] = rsqrtf(var + 1e-5f);
    }
    __syncthreads();
    const float mu = sres[0], inv = sres[1];
    F128 g  = *(const F128*)(gamma + c);
    F128 be = *(const F128*)(beta + c);
    float y0 = (x0 - mu) * inv * g.x + be.x;
    float y1 = (x1 - mu) * inv * g.y + be.y;
    float y2 = (x2 - mu) * inv * g.z + be.z;
    float y3 = (x3 - mu) * inv * g.w + be.w;
    if (out_f32) {
        F128 o; o.x = y0; o.y = y1; o.z = y2; o.w = y3;
        *(F128*)(out_f32 + (size_t)row * HID + c) = o;
    }
    if (out_bf16) {
        U64 o; o.x = pack2(y0, y1); o.y = pack2(y2, y3);
        *(U64*)(out_bf16 + (size_t)row * KCAT + INSZ + c) = o;
    }
}

// ---------------------------------------------------------------------------
// LSTM elementwise:  i,f,g,o -> cell = f*proc_c + i*g ; hidden = o*tanh(cell)
// d_out = [hidden (B*H) | cell (B*H)]
// ---------------------------------------------------------------------------
__global__ __launch_bounds__(256) void lstm_ew_kernel(const float* __restrict__ gates,
                                                      const float* __restrict__ procc,
                                                      const float* __restrict__ bih,
                                                      const float* __restrict__ bhh,
                                                      float* __restrict__ out) {
    const int j   = blockIdx.x * 256 + threadIdx.x;
    const int row = j >> 10, col = j & (HID - 1);
    const float* g = gates + (size_t)row * N4H;
    float vi = g[col]            + bih[col]            + bhh[col];
    float vf = g[col + HID]      + bih[col + HID]      + bhh[col + HID];
    float vg = g[col + 2 * HID]  + bih[col + 2 * HID]  + bhh[col + 2 * HID];
    float vo = g[col + 3 * HID]  + bih[col + 3 * HID]  + bhh[col + 3 * HID];
    vi = 1.0f / (1.0f + __expf(-vi));
    vf = 1.0f / (1.0f + __expf(-vf));
    vo = 1.0f / (1.0f + __expf(-vo));
    vg = tanhf(vg);
    float cell = vf * procc[j] + vi * vg;
    float hid  = vo * tanhf(cell);
    out[j] = hid;
    out[(size_t)BATCH * HID + j] = cell;
}

// ---------------------------------------------------------------------------
// Workspace layout (bytes).  gates aliases raw_h/raw_c (dead after LN).
// Total required: ~116 MB.
// ---------------------------------------------------------------------------
static constexpr size_t OFF_WH    = 0;                                   // bf16 1024x3072
static constexpr size_t OFF_WC    = OFF_WH   + (size_t)HID * K3H * 2;    //  6,291,456
static constexpr size_t OFF_WCAT  = OFF_WC   + (size_t)HID * K3H * 2;    // 12,582,912
static constexpr size_t OFF_XCAT  = OFF_WCAT + (size_t)N4H * KCAT * 2;   // 25,165,824
static constexpr size_t OFF_RAWH  = OFF_XCAT + (size_t)BATCH * KCAT * 2; // 37,748,736
static constexpr size_t OFF_RAWC  = OFF_RAWH + (size_t)BATCH * HID * 4;  // 54,525,952
static constexpr size_t OFF_GATES = OFF_RAWH;                            // aliases raws
static constexpr size_t OFF_PROCC = OFF_GATES + (size_t)BATCH * N4H * 4; // 104,857,600

extern "C" void kernel_launch(void* const* d_in, const int* in_sizes, int n_in,
                              void* d_out, int out_size, void* d_ws, size_t ws_size,
                              hipStream_t stream) {
    const float* x    = (const float*)d_in[0];
    const float* hl   = (const float*)d_in[1];
    const float* cl   = (const float*)d_in[2];
    const float* hu   = (const float*)d_in[3];
    const float* cu   = (const float*)d_in[4];
    const float* hp   = (const float*)d_in[5];
    const float* cp   = (const float*)d_in[6];
    const float* WhP  = (const float*)d_in[7];
    const float* bhP  = (const float*)d_in[8];
    const float* WcP  = (const float*)d_in[9];
    const float* bcP  = (const float*)d_in[10];
    const float* gamH = (const float*)d_in[11];
    const float* betH = (const float*)d_in[12];
    const float* gamC = (const float*)d_in[13];
    const float* betC = (const float*)d_in[14];
    const float* Wih  = (const float*)d_in[15];
    const float* Whh  = (const float*)d_in[16];
    const float* bih  = (const float*)d_in[17];
    const float* bhh  = (const float*)d_in[18];

    char* ws = (char*)d_ws;
    unsigned short* whb   = (unsigned short*)(ws + OFF_WH);
    unsigned short* wcb   = (unsigned short*)(ws + OFF_WC);
    unsigned short* wcat  = (unsigned short*)(ws + OFF_WCAT);
    unsigned short* xcat  = (unsigned short*)(ws + OFF_XCAT);
    float*          rawh  = (float*)(ws + OFF_RAWH);
    float*          rawc  = (float*)(ws + OFF_RAWC);
    float*          gates = (float*)(ws + OFF_GATES);
    float*          procc = (float*)(ws + OFF_PROCC);

    // 1) weight / input conversion to bf16
    f32_to_bf16_kernel<<<(HID * K3H) / 1024, 256, 0, stream>>>(WhP, whb, HID * K3H);
    f32_to_bf16_kernel<<<(HID * K3H) / 1024, 256, 0, stream>>>(WcP, wcb, HID * K3H);
    pack_wcat_kernel<<<((size_t)N4H * KCAT) / 1024, 256, 0, stream>>>(Wih, Whh, wcat);
    pack_x_kernel<<<((size_t)BATCH * INSZ) / 1024, 256, 0, stream>>>(x, xcat);

    // 2) processor GEMMs (WMMA bf16): raw = combined @ W^T
    dim3 g1(HID / BN, BATCH / BM);            // (8, 32)
    gemm_proc_kernel<<<g1, 256, 0, stream>>>(hl, hu, hp, whb, rawh);
    gemm_proc_kernel<<<g1, 256, 0, stream>>>(cl, cu, cp, wcb, rawc);

    // 3) LayerNorms: h -> bf16 into Xcat cols 512..1535 ; c -> fp32 proc_c
    ln_kernel<<<BATCH, 256, 0, stream>>>(rawh, bhP, gamH, betH, nullptr, xcat);
    ln_kernel<<<BATCH, 256, 0, stream>>>(rawc, bcP, gamC, betC, procc, nullptr);

    // 4) gates GEMM (fuses x@W_ih^T + proc_h@W_hh^T via K concatenation)
    dim3 g3(N4H / BN, BATCH / BM);            // (32, 32)
    gemm_gates_kernel<<<g3, 256, 0, stream>>>(xcat, wcat, gates);

    // 5) LSTM elementwise -> d_out = [hidden | cell]
    lstm_ew_kernel<<<((size_t)BATCH * HID) / 256, 256, 0, stream>>>(gates, procc, bih, bhh,
                                                                    (float*)d_out);
}
